// SimpleModel_35802847379916
// MI455X (gfx1250) — matrix operand order repaired and checked by reading, not compiled, not run on hardware
//
#include <hip/hip_runtime.h>

typedef _Float16 v16h __attribute__((ext_vector_type(16)));
typedef float    v8f  __attribute__((ext_vector_type(8)));

namespace {
constexpr int IN_ = 16;
constexpr int H1_ = 8;
constexpr int H2_ = 4;
constexpr int B_  = 4096;
constexpr int TILE = 16;              // batch rows per wave (WMMA M)
constexpr int WAVES_PER_BLOCK = 4;
}

__device__ __forceinline__ float fsig(float x) {
  // sigmoid(x) = 1 / (1 + 2^(-x*log2(e)))  -> v_exp_f32 + v_rcp_f32
  float e = __builtin_amdgcn_exp2f(x * -1.442695040888963f);
  return __builtin_amdgcn_rcpf(1.0f + e);
}
__device__ __forceinline__ float ftanhf(float x) {
  return 2.0f * fsig(2.0f * x) - 1.0f;
}

__global__ __launch_bounds__(128, 1) void lstm2_wmma(
    const float* __restrict__ x,
    const float* __restrict__ Wih1, const float* __restrict__ Whh1,
    const float* __restrict__ bih1, const float* __restrict__ bhh1,
    const float* __restrict__ Wih2, const float* __restrict__ Whh2,
    const float* __restrict__ bih2, const float* __restrict__ bhh2,
    const float* __restrict__ fc1_w, const float* __restrict__ fc1_b,
    const float* __restrict__ fc2_w, const float* __restrict__ fc2_b,
    float* __restrict__ out, int T)
{
  __shared__ float sh1[WAVES_PER_BLOCK][TILE * H1_];   // h1 transpose staging
  __shared__ float sh2[WAVES_PER_BLOCK][TILE * H2_];   // h2 transpose staging

  const int lane = threadIdx.x & 31;
  const int w    = threadIdx.x >> 5;
  const int tile = blockIdx.x * WAVES_PER_BLOCK + w;
  const int b0   = tile * TILE;
  const int col  = lane & 15;          // C-layout column / B-layout N
  const bool hi  = lane >= 16;

  // ---- Build f16 B operands once (K x N panels). Dense-f16 B layout (table order):
  // N=lane%16, h=lane/16: element e<8 -> K=8h+e ; element e>=8 -> K=16+8h+(e-8).
  v16h b1a, b1b, b2m;
  #pragma unroll
  for (int e = 0; e < 16; ++e) {
    float va, vb, v2;
    const int k = (e < 8) ? (hi ? 8 : 0) + e : 16 + (hi ? 8 : 0) + (e - 8);
    va = (k < IN_) ? Wih1[col * IN_ + k]
       : (k < IN_ + H1_) ? Whh1[col * H1_ + (k - IN_)] : 0.0f;
    vb = (k < IN_) ? Wih1[(col + 16) * IN_ + k]
       : (k < IN_ + H1_) ? Whh1[(col + 16) * H1_ + (k - IN_)] : 0.0f;
    v2 = (k < H1_)        ? Wih2[col * H1_ + k]
       : (k < H1_ + H2_)  ? Whh2[col * H2_ + (k - H1_)]
                          : 0.0f;
    b1a[e] = (_Float16)va;
    b1b[e] = (_Float16)vb;
    b2m[e] = (_Float16)v2;
  }

  // Bias preloaded into C accumulators (same value for every row of a column)
  const float biasA = bih1[col] + bhh1[col];
  const float biasB = bih1[col + 16] + bhh1[col + 16];
  const float bias2 = bih2[col] + bhh2[col];
  v8f cA, cB, c2b;
  #pragma unroll
  for (int e = 0; e < 8; ++e) { cA[e] = biasA; cB[e] = biasB; c2b[e] = bias2; }

  // Recurrent state (C-layout: vgpr r -> row, lane%16 -> hidden index)
  float c1[8], c2s[8];
  #pragma unroll
  for (int r = 0; r < 8; ++r) { c1[r] = 0.0f; c2s[r] = 0.0f; }
  _Float16 h1p[8] = {};                 // h1(t-1), A-layout packed (lanes 0-15)
  _Float16 h2p[4] = {};                 // h2(t-1), A-layout packed (lanes 16-31)

  const size_t rowBase = ((size_t)b0 + col) * IN_ + (hi ? 8 : 0);
  const size_t tStride = (size_t)B_ * IN_;

  float4 xa = *(const float4*)(x + rowBase);
  float4 xb = *(const float4*)(x + rowBase + 4);

  const bool jlt8 = (col < 8);
  const bool tmid = (col >= 8) && (col < 12);

  for (int t = 0; t < T; ++t) {
    // Software-pipeline next step's x
    float4 na = xa, nb = xb;
    if (t + 1 < T) {
      const float* p = x + rowBase + (size_t)(t + 1) * tStride;
      na = *(const float4*)(p);
      nb = *(const float4*)(p + 4);
    }
    if ((t & 15) == 0 && t + 16 < T)
      __builtin_prefetch(x + rowBase + (size_t)(t + 16) * tStride, 0, 0);

    // ---- Layer 1: A1 = [x(t) | h1(t-1) | pad]
    v16h a1;
    a1[0]=(_Float16)xa.x; a1[1]=(_Float16)xa.y; a1[2]=(_Float16)xa.z; a1[3]=(_Float16)xa.w;
    a1[4]=(_Float16)xb.x; a1[5]=(_Float16)xb.y; a1[6]=(_Float16)xb.z; a1[7]=(_Float16)xb.w;
    #pragma unroll
    for (int j = 0; j < 8; ++j) a1[8 + j] = hi ? (_Float16)0.0f : h1p[j];

    v8f accA = __builtin_amdgcn_wmma_f32_16x16x32_f16(false, a1, false, b1a, (short)0, cA, false, false);
    v8f accB = __builtin_amdgcn_wmma_f32_16x16x32_f16(false, a1, false, b1b, (short)0, cB, false, false);

    float h1v[8];
    #pragma unroll
    for (int r = 0; r < 8; ++r) {
      float sa = fsig(accA[r]);                    // sigmoid(i) cols<8, sigmoid(f) cols>=8
      float vv = accB[r];
      float sb = fsig(jlt8 ? 2.0f * vv : vv);
      sb = jlt8 ? 2.0f * sb - 1.0f : sb;           // tanh(g) cols<8, sigmoid(o) cols>=8
      float iv = sa;
      float fv = __shfl_xor(sa, 8, 32);
      float gv = sb;
      float ov = __shfl_xor(sb, 8, 32);
      c1[r]  = fv * c1[r] + iv * gv;
      h1v[r] = ov * ftanhf(c1[r]);
    }

    // ---- Transpose h1 (C-layout -> A-layout) through per-wave LDS tile
    if (jlt8) {
      const int mb = hi ? 8 : 0;
      #pragma unroll
      for (int r = 0; r < 8; ++r) sh1[w][(mb + r) * H1_ + col] = h1v[r];
    }
    __builtin_amdgcn_wave_barrier();               // DS in-order per wave
    _Float16 nh1p[8];
    if (!hi) {
      float4 p0 = *(const float4*)&sh1[w][lane * H1_];
      float4 p1 = *(const float4*)&sh1[w][lane * H1_ + 4];
      nh1p[0]=(_Float16)p0.x; nh1p[1]=(_Float16)p0.y; nh1p[2]=(_Float16)p0.z; nh1p[3]=(_Float16)p0.w;
      nh1p[4]=(_Float16)p1.x; nh1p[5]=(_Float16)p1.y; nh1p[6]=(_Float16)p1.z; nh1p[7]=(_Float16)p1.w;
    } else {
      #pragma unroll
      for (int j = 0; j < 8; ++j) nh1p[j] = (_Float16)0.0f;
    }

    // ---- Layer 2: A2 = [h1(t) | h2(t-1) | pad]
    v16h a2;
    #pragma unroll
    for (int e = 0; e < 8; ++e)
      a2[e] = hi ? ((e < H2_) ? h2p[e] : (_Float16)0.0f) : nh1p[e];
    #pragma unroll
    for (int e = 8; e < 16; ++e) a2[e] = (_Float16)0.0f;

    v8f acc2 = __builtin_amdgcn_wmma_f32_16x16x32_f16(false, a2, false, b2m, (short)0, c2b, false, false);

    float h2v[8];
    #pragma unroll
    for (int r = 0; r < 8; ++r) {
      float vv = acc2[r];
      float s2 = fsig(tmid ? 2.0f * vv : vv);
      s2 = tmid ? 2.0f * s2 - 1.0f : s2;           // tanh on cols 8..11 (g-gates)
      float i2 = s2;
      float f2 = __shfl_xor(s2, 4, 32);
      float g2 = __shfl_xor(s2, 8, 32);
      float o2 = __shfl_xor(s2, 12, 32);
      c2s[r] = f2 * c2s[r] + i2 * g2;
      h2v[r] = o2 * ftanhf(c2s[r]);
    }

    // ---- Transpose h2 through LDS
    if (col < H2_) {
      const int mb = hi ? 8 : 0;
      #pragma unroll
      for (int r = 0; r < 8; ++r) sh2[w][(mb + r) * H2_ + col] = h2v[r];
    }
    __builtin_amdgcn_wave_barrier();
    if (hi) {
      float4 q = *(const float4*)&sh2[w][(lane - 16) * H2_];
      h2p[0]=(_Float16)q.x; h2p[1]=(_Float16)q.y; h2p[2]=(_Float16)q.z; h2p[3]=(_Float16)q.w;
    }
    #pragma unroll
    for (int j = 0; j < 8; ++j) h1p[j] = nh1p[j];
    xa = na; xb = nb;
  }

  // ---- MLP head on h2(T-1): 4 -> relu(2) -> sigmoid(1)
  __builtin_amdgcn_wave_barrier();
  if (!hi) {
    float4 q = *(const float4*)&sh2[w][lane * H2_];
    float h3a = fmaxf(0.0f, q.x*fc1_w[0] + q.y*fc1_w[1] + q.z*fc1_w[2] + q.w*fc1_w[3] + fc1_b[0]);
    float h3b = fmaxf(0.0f, q.x*fc1_w[4] + q.y*fc1_w[5] + q.z*fc1_w[6] + q.w*fc1_w[7] + fc1_b[1]);
    out[b0 + lane] = fsig(h3a * fc2_w[0] + h3b * fc2_w[1] + fc2_b[0]);
  }
}

extern "C" void kernel_launch(void* const* d_in, const int* in_sizes, int n_in,
                              void* d_out, int out_size, void* d_ws, size_t ws_size,
                              hipStream_t stream) {
  (void)n_in; (void)out_size; (void)d_ws; (void)ws_size;
  const float* x     = (const float*)d_in[0];
  const float* Wih1  = (const float*)d_in[1];
  const float* Whh1  = (const float*)d_in[2];
  const float* bih1  = (const float*)d_in[3];
  const float* bhh1  = (const float*)d_in[4];
  const float* Wih2  = (const float*)d_in[5];
  const float* Whh2  = (const float*)d_in[6];
  const float* bih2  = (const float*)d_in[7];
  const float* bhh2  = (const float*)d_in[8];
  const float* fc1_w = (const float*)d_in[9];
  const float* fc1_b = (const float*)d_in[10];
  const float* fc2_w = (const float*)d_in[11];
  const float* fc2_b = (const float*)d_in[12];

  const int T = in_sizes[0] / (B_ * IN_);          // 512
  dim3 grid((B_ / TILE) / WAVES_PER_BLOCK);        // 64 blocks
  dim3 block(32 * WAVES_PER_BLOCK);                // 128 threads = 4 waves

  hipLaunchKernelGGL(lstm2_wmma, grid, block, 0, stream,
                     x, Wih1, Whh1, bih1, bhh1, Wih2, Whh2, bih2, bhh2,
                     fc1_w, fc1_b, fc2_w, fc2_b, (float*)d_out, T);
}
